// FirDownsample_33689723470082
// MI455X (gfx1250) — matrix operand order repaired and checked by reading, not compiled, not run on hardware
//
#include <hip/hip_runtime.h>

// ---------------------------------------------------------------------------
// Fused FIR(4x4, pad2) + 3x3 stride-2 conv  ==  6x6 stride-2 conv with
// weff = conv(w, flip(fir)).  Implicit GEMM over WMMA bf16 with hi/lo
// split-bf16 (3 products per K=32 step) for near-fp32 accuracy.
//   x:    (16, 320, 64, 64) f32
//   w:    (320, 320, 3, 3)  f32
//   fir:  (4, 4)            f32
//   out:  (16, 320, 32, 32) f32
// Wave tiling: each wave computes 32co x 16ow (2 accumulators) so one
// B-fragment pair feeds 6 WMMAs -> LDS issue rate halved vs 16x16/wave.
// ---------------------------------------------------------------------------

typedef __attribute__((ext_vector_type(16))) __bf16 v16bf;
typedef __attribute__((ext_vector_type(8)))  float  v8f;

#define C_CH   320
#define N_IMG  16
#define H_IN   64
#define W_IN   64
#define OHI    32
#define OWI    32
#define TAPS   36      // 6x6 effective kernel
#define CB     32      // ci per K-block (one WMMA K step)
#define NCB    (C_CH / CB)   // 10
#define COB    64      // co per workgroup
#define SROWS  6
#define SCOLS  68      // 2*31 + 5 + 1
#define BLOCK_THREADS 128    // 4 waves

union FragBf {
    v16bf v;
    uint4 q[2];
};

__device__ __forceinline__ unsigned short f32_to_bf16_rne(float f) {
    unsigned int u = __float_as_uint(f);
    unsigned int r = u + 0x7FFFu + ((u >> 16) & 1u);   // round-to-nearest-even
    return (unsigned short)(r >> 16);
}
__device__ __forceinline__ float bf16_bits_to_f32(unsigned short h) {
    return __uint_as_float(((unsigned int)h) << 16);
}

// ---------------------------------------------------------------------------
// Stage A: build effective 6x6 kernel, split into bf16 hi/lo, layout [co][t][ci]
// ---------------------------------------------------------------------------
__global__ __launch_bounds__(256)
void weff_kernel(const float* __restrict__ w,
                 const float* __restrict__ fir,
                 unsigned short* __restrict__ whi,
                 unsigned short* __restrict__ wlo) {
    int idx = blockIdx.x * 256 + threadIdx.x;          // < 320*36*320 = 3,686,400
    int ci = idx % C_CH;
    int t  = (idx / C_CH) % TAPS;
    int co = idx / (C_CH * TAPS);
    int r = t / 6, s = t % 6;
    float acc = 0.0f;
    #pragma unroll
    for (int kh = 0; kh < 3; ++kh) {
        int fh = r - kh;
        if (fh < 0 || fh > 3) continue;
        #pragma unroll
        for (int kw = 0; kw < 3; ++kw) {
            int fw = s - kw;
            if (fw < 0 || fw > 3) continue;
            // kf = flip(fir): kf[fh][fw] = fir[3-fh][3-fw]
            acc += w[((co * C_CH + ci) * 3 + kh) * 3 + kw] * fir[(3 - fh) * 4 + (3 - fw)];
        }
    }
    unsigned short h = f32_to_bf16_rne(acc);
    unsigned short l = f32_to_bf16_rne(acc - bf16_bits_to_f32(h));
    whi[idx] = h;
    wlo[idx] = l;
}

// ---------------------------------------------------------------------------
// Stage B: main conv.  Grid (5 co-blocks, 32 oh, 16 n), 128 threads = 4 waves.
// Wave w: owSub = w&1 (16 ow), coPair = w>>1 (32 co -> two 16-co subtiles).
// K loop: 10 ci-blocks x 36 taps; per tap: 1 B hi/lo pair + 2 A hi/lo pairs,
// 6 WMMAs (hi*hi, hi*lo, lo*hi per accumulator).
// ---------------------------------------------------------------------------
__global__ __launch_bounds__(BLOCK_THREADS)
void conv6x6_wmma_kernel(const float* __restrict__ x,
                         const unsigned short* __restrict__ whi,
                         const unsigned short* __restrict__ wlo,
                         float* __restrict__ out) {
    // Input slab for this (n, oh): 6 rows x 68 cols x 32 ci, ci innermost
    __shared__ __attribute__((aligned(16))) unsigned short shi[SROWS * SCOLS * CB];
    __shared__ __attribute__((aligned(16))) unsigned short slo[SROWS * SCOLS * CB];

    const int coBlk = blockIdx.x;      // 0..4
    const int oh    = blockIdx.y;      // 0..31
    const int nImg  = blockIdx.z;      // 0..15

    const int tid    = threadIdx.x;
    const int lane   = tid & 31;
    const int wave   = tid >> 5;       // 0..3
    const int owSub  = wave & 1;       // 0..1
    const int coPair = wave >> 1;      // 0..1  -> co subtiles coPair*2, coPair*2+1
    const int laneLo = lane & 15;
    const int half   = lane >> 4;      // 0: lanes 0-15, 1: lanes 16-31

    const int coBase  = coBlk * COB;
    const int coRow0  = coBase + coPair * 32 + laneLo;   // A row M, subtile 0
    const int coRow1  = coRow0 + 16;                     // A row M, subtile 1
    const int myOw    = owSub * 16 + laneLo;             // B col N
    const int ihBase  = 2 * oh - 2;

    // Per-lane fragment chunk offsets (elements) per documented VGPR striping.
    // A (16x32, 16-bit): lanes 0-15 hold K 0..7 & 16..23; lanes 16-31: 8..15 & 24..31
    const int aOff = half * 8;          // chunk0 at aOff, chunk1 at aOff+16
    // B (32x16, 16-bit): lanes 0-15 hold K 0..7 & 16..23; lanes 16-31: 8..15 & 24..31
    const int bOff = half * 8;          // chunk0 at bOff, chunk1 at bOff+16

    v8f acc0 = {};
    v8f acc1 = {};

    // Slab-fill mapping with no div/mod: 32 ci groups x 4 lanes, 6 rows x 17 col
    // steps each (68 cols / 4) -> exactly 102 elems/thread, no divergence.
    const int fillCi   = tid >> 2;      // 0..31
    const int fillLane = tid & 3;       // 0..3

    for (int cb = 0; cb < NCB; ++cb) {
        const int ciBase = cb * CB;

        __syncthreads();
        // ---- stage slab: global f32 -> LDS bf16 hi/lo, ci innermost
        {
            const long ciRowBase = ((long)nImg * C_CH + (ciBase + fillCi)) * H_IN;
            #pragma unroll
            for (int row = 0; row < SROWS; ++row) {
                const int  ih     = ihBase + row;
                const bool rowOk  = (ih >= 0) && (ih < H_IN);
                const long gRow   = (ciRowBase + ih) * W_IN;
                for (int c = 0; c < 17; ++c) {
                    const int col = fillLane + c * 4;     // 0..67
                    const int iw  = col - 2;
                    float v = 0.0f;
                    if (rowOk && iw >= 0 && iw < W_IN) {
                        v = x[gRow + iw];
                        // pull next ci-block's line into cache while we compute
                        if (cb + 1 < NCB)
                            __builtin_prefetch(&x[gRow + (long)CB * H_IN * W_IN + iw], 0, 0);
                    }
                    unsigned short h = f32_to_bf16_rne(v);
                    unsigned short l = f32_to_bf16_rne(v - bf16_bits_to_f32(h));
                    int di = (row * SCOLS + col) * CB + fillCi;
                    shi[di] = h;
                    slo[di] = l;
                }
            }
        }
        __syncthreads();

        // ---- 36 taps: each is two 16x16x32 GEMM steps sharing one B pair
        for (int r = 0; r < 6; ++r) {
            #pragma unroll
            for (int s = 0; s < 6; ++s) {
                const int t = r * 6 + s;

                // B fragments from LDS slab: col = 2*ow + s (stride-2 gather)
                const int col  = 2 * myOw + s;                     // 0..67
                const int base = (r * SCOLS + col) * CB + bOff;    // 16B aligned
                FragBf b_hi, b_lo;
                b_hi.q[0] = *(const uint4*)(&shi[base]);
                b_hi.q[1] = *(const uint4*)(&shi[base + 16]);
                b_lo.q[0] = *(const uint4*)(&slo[base]);
                b_lo.q[1] = *(const uint4*)(&slo[base + 16]);

                // A fragments (weff hi/lo) from global (L1/L2 hot), 2 co subtiles
                const unsigned short* aH0 =
                    whi + ((long)coRow0 * TAPS + t) * C_CH + ciBase + aOff;
                const unsigned short* aL0 =
                    wlo + ((long)coRow0 * TAPS + t) * C_CH + ciBase + aOff;
                const unsigned short* aH1 =
                    whi + ((long)coRow1 * TAPS + t) * C_CH + ciBase + aOff;
                const unsigned short* aL1 =
                    wlo + ((long)coRow1 * TAPS + t) * C_CH + ciBase + aOff;
                FragBf a_hi0, a_lo0, a_hi1, a_lo1;
                a_hi0.q[0] = *(const uint4*)(aH0);
                a_hi0.q[1] = *(const uint4*)(aH0 + 16);
                a_lo0.q[0] = *(const uint4*)(aL0);
                a_lo0.q[1] = *(const uint4*)(aL0 + 16);
                a_hi1.q[0] = *(const uint4*)(aH1);
                a_hi1.q[1] = *(const uint4*)(aH1 + 16);
                a_lo1.q[0] = *(const uint4*)(aL1);
                a_lo1.q[1] = *(const uint4*)(aL1 + 16);

                // split-bf16: a*b ~= ah*bh + ah*bl + al*bh (f32 accumulate)
                acc0 = __builtin_amdgcn_wmma_f32_16x16x32_bf16(
                    false, a_hi0.v, false, b_hi.v, (short)0, acc0, false, false);
                acc1 = __builtin_amdgcn_wmma_f32_16x16x32_bf16(
                    false, a_hi1.v, false, b_hi.v, (short)0, acc1, false, false);
                acc0 = __builtin_amdgcn_wmma_f32_16x16x32_bf16(
                    false, a_hi0.v, false, b_lo.v, (short)0, acc0, false, false);
                acc1 = __builtin_amdgcn_wmma_f32_16x16x32_bf16(
                    false, a_hi1.v, false, b_lo.v, (short)0, acc1, false, false);
                acc0 = __builtin_amdgcn_wmma_f32_16x16x32_bf16(
                    false, a_lo0.v, false, b_hi.v, (short)0, acc0, false, false);
                acc1 = __builtin_amdgcn_wmma_f32_16x16x32_bf16(
                    false, a_lo1.v, false, b_hi.v, (short)0, acc1, false, false);
            }
        }
    }

    // ---- epilogue: documented 16x16 f32 C/D layout
    // VGPR rr: M = rr + (lane>=16 ? 8 : 0), N = lane & 15
    const int mBase0 = coBase + coPair * 32 + half * 8;
    const int mBase1 = mBase0 + 16;
    const int owOut  = owSub * 16 + laneLo;
    #pragma unroll
    for (int rr = 0; rr < 8; ++rr) {
        out[(((long)nImg * C_CH + (mBase0 + rr)) * OHI + oh) * OWI + owOut] = acc0[rr];
        out[(((long)nImg * C_CH + (mBase1 + rr)) * OHI + oh) * OWI + owOut] = acc1[rr];
    }
}

// ---------------------------------------------------------------------------
extern "C" void kernel_launch(void* const* d_in, const int* in_sizes, int n_in,
                              void* d_out, int out_size, void* d_ws, size_t ws_size,
                              hipStream_t stream) {
    const float* x   = (const float*)d_in[0];   // 16*320*64*64
    const float* w   = (const float*)d_in[1];   // 320*320*3*3
    const float* fir = (const float*)d_in[2];   // 4*4 (already normalized)
    float* out = (float*)d_out;                 // 16*320*32*32

    const size_t weffElems = (size_t)C_CH * TAPS * C_CH;   // 3,686,400
    unsigned short* whi = (unsigned short*)d_ws;
    unsigned short* wlo = whi + weffElems;                 // ~14.8 MB total

    // Stage A: effective 6x6 kernel, split bf16
    {
        int total  = (int)weffElems;               // 3,686,400 = 14400 * 256
        int blocks = (total + 255) / 256;
        weff_kernel<<<blocks, 256, 0, stream>>>(w, fir, whi, wlo);
    }

    // Stage B: WMMA conv
    {
        dim3 grid(C_CH / COB, OHI, N_IMG);         // (5, 32, 16)
        conv6x6_wmma_kernel<<<grid, BLOCK_THREADS, 0, stream>>>(x, whi, wlo, out);
    }
}
